// LaEnergiaNoAparece_38371237822694
// MI455X (gfx1250) — matrix operand order repaired and checked by reading, not compiled, not run on hardware
//
#include <hip/hip_runtime.h>
#include <hip/hip_bf16.h>

// CDNA5 / gfx1250: wave32, WMMA 16x16x32 bf16.
typedef __attribute__((ext_vector_type(16))) __bf16 v16bf;
typedef __attribute__((ext_vector_type(8)))  __bf16 v8bf;
typedef __attribute__((ext_vector_type(8)))  float  v8f;

#define NNODES 169          // 13*13
#define KPAD   192          // 6 * 32 (k-dim padded for WMMA steps)
#define NROWS  170          // 169 real rows + 1 dummy all-zero row (branch-free pad)
#define NT     11           // ceil(169/16) N tiles
#define KS     6            // ceil(169/32) K steps
#define BOLTZ  5.67e-8f
// N-tiles containing heater/interface nodes: nt in {0,2,5,7,9,10}
#define QTILEMASK 0x6A5u

__device__ __forceinline__ v16bf load_bfrag(const __bf16* rowK, int ks) {
    const v8bf* p = (const v8bf*)(rowK + ks * 32);
    v8bf lo = p[0], hi = p[1];
    v16bf bf;
    #pragma unroll
    for (int e = 0; e < 8; ++e) { bf[e] = lo[e]; bf[e + 8] = hi[e]; }
    return bf;
}

__global__ __launch_bounds__(256)
void LaEnergia_thermal_wmma(const float* __restrict__ outputs,
                            const float* __restrict__ heaters,
                            const float* __restrict__ interfaces,
                            const float* __restrict__ Tenv,
                            const float* __restrict__ Kg,
                            const float* __restrict__ Ed,
                            float* __restrict__ out,
                            int Btot, int numTiles, float invN)
{
    // B-operand (K^T) resident in LDS as bf16: Kb[n][k], k contiguous so a
    // lane's 16 k-values are two aligned ds_load_b128. Row 169 is all zeros so
    // out-of-range N lanes clamp to it. Columns k>=169 are zero so the A
    // operand never needs masking (garbage * 0 == 0). 170*192*2 = 65,280 B.
    __shared__ __bf16 Kb[NROWS * KPAD];
    // Per-node Q code: 0=none, 1..4=heater slot+1, 5..8=interface slot+5.
    __shared__ unsigned char qtab[NNODES];

    for (int idx = threadIdx.x; idx < NROWS * KPAD; idx += blockDim.x) {
        int n = idx / KPAD;
        int k = idx - n * KPAD;
        float v = (n < NNODES && k < NNODES) ? Kg[n * NNODES + k] : 0.0f;
        Kb[idx] = (__bf16)v;                  // B[k][n] = K[n][k]
    }
    for (int idx = threadIdx.x; idx < NNODES; idx += blockDim.x) {
        int code = 0;
        if      (idx ==  81) code = 1;
        else if (idx ==  45) code = 2;
        else if (idx == 120) code = 3;
        else if (idx == 126) code = 4;
        else if (idx ==   0) code = 5;
        else if (idx ==  12) code = 6;
        else if (idx == 168) code = 7;
        else if (idx == 156) code = 8;
        qtab[idx] = (unsigned char)code;
    }
    __syncthreads();

    const int lane   = threadIdx.x & 31;
    const int waveIn = threadIdx.x >> 5;
    const int wavesPerBlock = blockDim.x >> 5;
    const int gwave  = blockIdx.x * wavesPerBlock + waveIn;
    const int nwaves = gridDim.x * wavesPerBlock;
    const int laneLo = lane & 15;
    const int hiHalf = (lane < 16) ? 0 : 1;   // upper half-wave flag

    float lsum = 0.0f;

    // persistent waves: each wave owns 16-batch tiles, grid-stride
    for (int tile = gwave; tile < numTiles; tile += nwaves) {
        const int b0 = tile * 16;
        const int bb = b0 + hiHalf * 8;            // M base for this half-wave
        const bool fullTile = (b0 + 16 <= Btot);

        // ---- A fragments: 16 rows of T (bf16), contiguous k = 16*half + e (same order as B).
        //      Unconditional loads; k>=169 uses clamped address (B side is 0).
        int m = b0 + laneLo;
        if (m >= Btot) m = Btot - 1;               // safe clamp (tail masked below)
        const unsigned aoff = (unsigned)m * (unsigned)NNODES; // u32 idx -> saddr+voffset
        v16bf afr[KS];
        #pragma unroll
        for (int ks = 0; ks < KS; ++ks) {
            const int k0 = ks * 32 + hiHalf * 16;
            v16bf a;
            #pragma unroll
            for (int e = 0; e < 16; ++e) {
                int k  = k0 + e;
                int kc = (k < NNODES) ? k : (NNODES - 1);   // folds for ks<5
                a[e] = (__bf16)outputs[aoff + (unsigned)kc];
            }
            afr[ks] = a;
        }

        // ---- 11 N-tiles: 12 ds_load_b128 + 6 WMMAs, fused epilogue ----
        for (int nt = 0; nt < NT; ++nt) {
            const int n  = nt * 16 + laneLo;          // D: N = lane%16
            const int rn = (n < NNODES) ? n : NNODES; // clamp to dummy zero row
            const __bf16* rowK = &Kb[rn * KPAD + hiHalf * 16];

            v16bf bfr[KS];
            #pragma unroll
            for (int ks = 0; ks < KS; ++ks) bfr[ks] = load_bfrag(rowK, ks);

            v8f acc = {};
            #pragma unroll
            for (int ks = 0; ks < KS; ++ks) {
                acc = __builtin_amdgcn_wmma_f32_16x16x32_bf16(
                          false, afr[ks], false, bfr[ks], (short)0, acc, false, false);
            }

            // ---- fused epilogue: u32 offsets -> saddr+voffset addressing ----
            const bool nValid = (n < NNODES);
            const unsigned nc = (unsigned)(nValid ? n : 0); // safe addr, masked later

            const float EB = BOLTZ * Ed[nc];

            float exv[8];
            {
                float Ts[8], Tes[8];
                #pragma unroll
                for (int v = 0; v < 8; ++v) {
                    int b  = bb + v;                  // D: M = vgpr (+8 hi half)
                    int bc = (b < Btot) ? b : (Btot - 1);
                    const unsigned off = (unsigned)bc * (unsigned)NNODES + nc;
                    Ts[v]  = outputs[off];            // L2-hot, batched, 32-bit voffset
                    Tes[v] = Tenv[off];
                }
                #pragma unroll
                for (int v = 0; v < 8; ++v) {
                    const float T  = Ts[v],  Te = Tes[v];
                    const float T2 = T * T,  Te2 = Te * Te;
                    exv[v] = acc[v] + EB * (T2 * T2 - Te2 * Te2);
                }
            }

            // Q only exists in tiles {0,2,5,7,9,10}: uniform scalar branch,
            // non-Q tiles carry zero extra instructions.
            if ((QTILEMASK >> nt) & 1u) {
                const int  code = qtab[nc];
                const bool hasQ = (code != 0);
                const int  slot = (code - 1) & 3;
                const float* qbase = (code <= 4) ? heaters : interfaces;
                #pragma unroll
                for (int v = 0; v < 8; ++v) {
                    int b  = bb + v;
                    int bc = (b < Btot) ? b : (Btot - 1);
                    float qv = qbase[(unsigned)bc * 4u + (unsigned)slot];
                    exv[v] -= hasQ ? qv : 0.0f;
                }
            }

            #pragma unroll
            for (int v = 0; v < 8; ++v) {
                const bool valid = nValid && (fullTile || (bb + v < Btot));
                lsum += valid ? fabsf(exv[v]) : 0.0f; // select, no exec divergence
            }
        }
    }

    // wave32 shuffle reduction, one f32 atomic per wave (pre-scaled -> mean)
    #pragma unroll
    for (int offs = 16; offs > 0; offs >>= 1)
        lsum += __shfl_down(lsum, offs, 32);
    if (lane == 0)
        atomicAdd(out, lsum * invN);
}

extern "C" void kernel_launch(void* const* d_in, const int* in_sizes, int n_in,
                              void* d_out, int out_size, void* d_ws, size_t ws_size,
                              hipStream_t stream) {
    const float* outputs    = (const float*)d_in[0];   // [B,13,13]
    const float* heaters    = (const float*)d_in[1];   // [B,4]
    const float* interfaces = (const float*)d_in[2];   // [B,4]
    const float* Tenv       = (const float*)d_in[3];   // [B,169]
    const float* Kg         = (const float*)d_in[4];   // [169,169]
    const float* Ed         = (const float*)d_in[5];   // [169]

    const int B = in_sizes[0] / NNODES;
    const int numTiles = (B + 15) / 16;
    const float invN = 1.0f / ((float)B * (float)NNODES);

    // scalar accumulator must start at zero (harness poisons d_out)
    hipMemsetAsync(d_out, 0, (size_t)out_size * sizeof(float), stream);

    const int threads = 256;                 // 8 waves / workgroup
    int grid = (numTiles + 7) / 8;           // one tile per wave minimum
    if (grid > 1024) grid = 1024;            // persistent waves, grid-stride
    if (grid < 1) grid = 1;

    LaEnergia_thermal_wmma<<<grid, threads, 0, stream>>>(
        outputs, heaters, interfaces, Tenv, Kg, Ed,
        (float*)d_out, B, numTiles, invN);
}
